// AMD_Layer_20160576488013
// MI455X (gfx1250) — compile-verified, matrix operand order checked by reading, not run on hardware
//
#include <hip/hip_runtime.h>
#include <hip/hip_bf16.h>

#define B_      2
#define N_      16384
#define DIM_    512
#define H_      8
#define A_      256
#define D_      64
#define BH_     16
#define NCHUNK_ 32
#define CHUNK_  512

typedef __attribute__((ext_vector_type(16))) __bf16 v16bf;
typedef __attribute__((ext_vector_type(8)))  __bf16 v8bf;
typedef __attribute__((ext_vector_type(8)))  float  v8f;
typedef __attribute__((ext_vector_type(4)))  unsigned int v4u;
typedef __attribute__((ext_vector_type(8)))  int v8i;
typedef __attribute__((ext_vector_type(4)))  int v4i;

static __device__ __forceinline__ v8f wmma_bf16(v16bf a, v16bf b, v8f c) {
  return __builtin_amdgcn_wmma_f32_16x16x32_bf16(false, a, false, b, (short)0, c,
                                                 false, false);
}

// Load a 16x32 (MxK) row-major bf16 tile into an A/B^T fragment.
// ISA layout: lane<16 holds row (lane&15) K=0..7 and 16..23;
//             lane>=16 holds same row K=8..15 and 24..31.
static __device__ __forceinline__ v16bf frag_ld(const __bf16* tile, int ld, int lane) {
  int r = lane & 15, hl = lane >> 4;
  const __bf16* rp = tile + (size_t)r * ld + hl * 8;
  v8bf lo = *(const v8bf*)(rp);
  v8bf hi = *(const v8bf*)(rp + 16);
  return __builtin_shufflevector(lo, hi, 0, 1, 2, 3, 4, 5, 6, 7,
                                 8, 9, 10, 11, 12, 13, 14, 15);
}

static __device__ __forceinline__ float sigmoidf_(float x) {
  return 1.f / (1.f + __expf(-x));
}

// TDM 2-D tile load: bf16 elements, tile == tensor window (no OOB),
// rows of tile_d0 elements, tile_d1 rows, source row stride stride0 (elements).
static __device__ __forceinline__ void tdm_load_2d(unsigned lds_addr, const void* gptr,
                                                   unsigned tile_d0, unsigned tile_d1,
                                                   unsigned stride0) {
  unsigned long long ga = (unsigned long long)(size_t)gptr;
  v4u g0;
  g0[0] = 1u;                                           // count=1, user mode
  g0[1] = lds_addr;                                     // LDS byte address
  g0[2] = (unsigned)ga;                                 // global_addr[31:0]
  g0[3] = (unsigned)((ga >> 32) & 0x1FFFFFFu) | (2u << 30);  // addr[56:32] | type=2
  v8i g1;
  g1[0] = (int)(1u << 16);        // workgroup_mask=0, data_size=1 (2 bytes)
  g1[1] = (int)(tile_d0 << 16);   // tensor_dim0[15:0] in [31:16]
  g1[2] = (int)(tile_d1 << 16);   // tensor_dim0[31:16]=0 | tensor_dim1[15:0]
  g1[3] = (int)(tile_d0 << 16);   // tensor_dim1[31:16]=0 | tile_dim0
  g1[4] = (int)tile_d1;           // tile_dim1 | tile_dim2=0
  g1[5] = (int)stride0;           // tensor_dim0_stride[31:0]
  g1[6] = 0;                      // stride hi | tensor_dim1_stride lo
  g1[7] = 0;
  v4i z4 = {0, 0, 0, 0};
  v8i z8 = {0, 0, 0, 0, 0, 0, 0, 0};
  __builtin_amdgcn_tensor_load_to_lds(g0, g1, z4, z4, z8, 0);
}

// ---------------- Kernel 0: convert w_qkv and agent to bf16 ----------------
__global__ void cvt_weights(const float* __restrict__ wq,
                            const float* __restrict__ agent,
                            __bf16* __restrict__ wqb, __bf16* __restrict__ agb) {
  int idx = blockIdx.x * 256 + threadIdx.x;
  if (idx < 3 * DIM_ * DIM_) wqb[idx] = (__bf16)wq[idx];
  int i2 = idx - 3 * DIM_ * DIM_;
  if (i2 >= 0 && i2 < H_ * A_ * D_) agb[i2] = (__bf16)agent[i2];
}

// ---------------- Kernel 1: QKV GEMM (bf16 WMMA), head split ----------------
// out[m,o] = sum_k x[m,k] * w_qkv[o,k];  q,k -> [bh,n,d] bf16, v -> [bh,d,n] bf16
// grid 2048 blocks; K outer / o-groups inner with 12 live accumulators so the
// LDS A-fragments are loaded exactly once (no LICM spills).
__global__ void qkv_gemm(const float* __restrict__ x, const __bf16* __restrict__ wqb,
                         __bf16* __restrict__ qb, __bf16* __restrict__ kb,
                         __bf16* __restrict__ vtb) {
  __shared__ __attribute__((aligned(32))) __bf16 xs[16][512];
  int mtile = blockIdx.x;
  int tid = threadIdx.x;
  for (int e = tid; e < 16 * 512; e += 256) {
    int rr = e >> 9, c = e & 511;
    xs[rr][c] = (__bf16)x[((size_t)mtile * 16 + rr) * 512 + c];
  }
  __syncthreads();
  int wave = tid >> 5, lane = tid & 31, r = lane & 15, hl = lane >> 4;
  const __bf16* wb0 = wqb + (size_t)(wave * 16) * 512;
  v8f zero = {0.f, 0.f, 0.f, 0.f, 0.f, 0.f, 0.f, 0.f};
  v8f acc[12];
#pragma unroll
  for (int og = 0; og < 12; ++og) acc[og] = zero;
#pragma unroll
  for (int ks = 0; ks < 16; ++ks) {
    v16bf a = frag_ld(&xs[0][0] + ks * 32, 512, lane);
#pragma unroll
    for (int og = 0; og < 12; ++og) {
      v16bf bb = frag_ld(wb0 + (size_t)og * 128 * 512 + ks * 32, 512, lane);
      acc[og] = wmma_bf16(a, bb, acc[og]);
    }
  }
#pragma unroll
  for (int og = 0; og < 12; ++og) {
    int obase = og * 128 + wave * 16;
    int o = obase + r;
    int which = o >> 9, oo = o & 511, hh = oo >> 6, d = oo & 63;
#pragma unroll
    for (int rr2 = 0; rr2 < 8; ++rr2) {
      int m = mtile * 16 + rr2 + 8 * hl;
      int b = m >> 14, n = m & (N_ - 1);
      int bh = b * 8 + hh;
      __bf16 val = (__bf16)acc[og][rr2];
      if (which == 0)      qb[((size_t)bh * N_ + n) * 64 + d] = val;
      else if (which == 1) kb[((size_t)bh * N_ + n) * 64 + d] = val;
      else                 vtb[((size_t)bh * 64 + d) * N_ + n] = val;
    }
  }
}

// -------- Kernel 3: flash-style softmax_n(agent @ k^T) @ v partials --------
// grid (NCHUNK_, BH_), block 512 (16 waves, one 16-row agent tile per wave).
// k/v tiles staged in LDS by the Tensor Data Mover, double-buffered.
__global__ void flash_kv(const __bf16* __restrict__ agb, const __bf16* __restrict__ kb_all,
                         const __bf16* __restrict__ vtb_all,
                         float* __restrict__ Pm, float* __restrict__ Pl,
                         float* __restrict__ Pkv) {
  __shared__ __attribute__((aligned(128))) __bf16 ks_[2][32][64];   // k rows (n,d)
  __shared__ __attribute__((aligned(128))) __bf16 vs_[2][64][32];   // v^T rows (d,n)
  __shared__ __attribute__((aligned(32)))  __bf16 pstage[16][16][32];
  int chunk = blockIdx.x, bh = blockIdx.y, h = bh & 7;
  int tid = threadIdx.x, wave = tid >> 5, lane = tid & 31, r = lane & 15, hl = lane >> 4;
  int abase = wave * 16;
  const __bf16* ag = agb + ((size_t)h * A_ + abase) * D_;
  const __bf16* kbase = kb_all + (size_t)bh * N_ * D_;
  const __bf16* vbase = vtb_all + (size_t)bh * D_ * N_;
  int n0 = chunk * CHUNK_;

  if (wave == 0) {  // TDM issue for step 0 (EXEC-independent, one per wave)
    tdm_load_2d((unsigned)(size_t)&ks_[0][0][0], kbase + (size_t)n0 * 64, 64, 32, 64);
    tdm_load_2d((unsigned)(size_t)&vs_[0][0][0], vbase + n0, 32, 64, N_);
  }

  v16bf aA0 = frag_ld(ag, 64, lane);
  v16bf aA1 = frag_ld(ag + 32, 64, lane);
  v8f zero = {0.f, 0.f, 0.f, 0.f, 0.f, 0.f, 0.f, 0.f};
  v8f acc[4];
#pragma unroll
  for (int q = 0; q < 4; ++q) acc[q] = zero;
  float m_[8], l_[8];
#pragma unroll
  for (int rr = 0; rr < 8; ++rr) { m_[rr] = -1e30f; l_[rr] = 0.f; }

  for (int step = 0; step < CHUNK_ / 32; ++step) {
    int cur = step & 1;
    if (wave == 0) {
      if (step + 1 < CHUNK_ / 32) {
        int nb1 = n0 + (step + 1) * 32;
        tdm_load_2d((unsigned)(size_t)&ks_[cur ^ 1][0][0], kbase + (size_t)nb1 * 64, 64, 32, 64);
        tdm_load_2d((unsigned)(size_t)&vs_[cur ^ 1][0][0], vbase + nb1, 32, 64, N_);
        __builtin_amdgcn_s_wait_tensorcnt((short)2);  // current buffers landed
      } else {
        __builtin_amdgcn_s_wait_tensorcnt((short)0);
      }
    }
    __syncthreads();
    // scores: two 16x16 tiles over the 32 staged n-rows (K = D = 64)
    v16bf b0 = frag_ld(&ks_[cur][0][0], 64, lane);
    v16bf b1 = frag_ld(&ks_[cur][0][0] + 32, 64, lane);
    v8f s0 = zero;
    s0 = wmma_bf16(aA0, b0, s0);
    s0 = wmma_bf16(aA1, b1, s0);
    b0 = frag_ld(&ks_[cur][16][0], 64, lane);
    b1 = frag_ld(&ks_[cur][16][0] + 32, 64, lane);
    v8f s1 = zero;
    s1 = wmma_bf16(aA0, b0, s1);
    s1 = wmma_bf16(aA1, b1, s1);
    // online softmax in registers: row m = rr + 8*hl lives in one VGPR
    // across a 16-lane half -> xor-shuffle reductions.
    float f8[8];
#pragma unroll
    for (int rr = 0; rr < 8; ++rr) {
      float vmax = fmaxf(s0[rr], s1[rr]);
      vmax = fmaxf(vmax, __shfl_xor(vmax, 1, 32));
      vmax = fmaxf(vmax, __shfl_xor(vmax, 2, 32));
      vmax = fmaxf(vmax, __shfl_xor(vmax, 4, 32));
      vmax = fmaxf(vmax, __shfl_xor(vmax, 8, 32));
      float mn = fmaxf(m_[rr], vmax);
      float f = __expf(m_[rr] - mn);
      float p0 = __expf(s0[rr] - mn);
      float p1 = __expf(s1[rr] - mn);
      float rs = p0 + p1;
      rs += __shfl_xor(rs, 1, 32);
      rs += __shfl_xor(rs, 2, 32);
      rs += __shfl_xor(rs, 4, 32);
      rs += __shfl_xor(rs, 8, 32);
      l_[rr] = l_[rr] * f + rs;
      m_[rr] = mn;
      f8[rr] = f;
      pstage[wave][rr + 8 * hl][r] = (__bf16)p0;
      pstage[wave][rr + 8 * hl][16 + r] = (__bf16)p1;
    }
    asm volatile("s_wait_dscnt 0x0" ::: "memory");
#pragma unroll
    for (int q = 0; q < 4; ++q)
#pragma unroll
      for (int rr = 0; rr < 8; ++rr) acc[q][rr] *= f8[rr];
    v16bf pA = frag_ld(&pstage[wave][0][0], 32, lane);
#pragma unroll
    for (int q = 0; q < 4; ++q) {
      v16bf bv = frag_ld(&vs_[cur][q * 16][0], 32, lane);
      acc[q] = wmma_bf16(pA, bv, acc[q]);
    }
    __syncthreads();  // all waves done with buffers before TDM refills them
  }

  size_t pbase = ((size_t)bh * NCHUNK_ + chunk) * A_ + abase;
  if (r == 0) {  // lane 0 owns rows 0..7, lane 16 rows 8..15 (state replicated)
#pragma unroll
    for (int rr = 0; rr < 8; ++rr) {
      Pm[pbase + rr + 8 * hl] = m_[rr];
      Pl[pbase + rr + 8 * hl] = l_[rr];
    }
  }
#pragma unroll
  for (int q = 0; q < 4; ++q)
#pragma unroll
    for (int rr = 0; rr < 8; ++rr)
      Pkv[(pbase + rr + 8 * hl) * D_ + q * 16 + r] = acc[q][rr];
}

// -------- Kernel 4a: merge partials -> kv [bh,a,d], + threshold partial sums --------
__global__ void merge_kv(const float* __restrict__ Pm, const float* __restrict__ Pl,
                         const float* __restrict__ Pkv, const float* __restrict__ w_thresh,
                         float* __restrict__ kv, float* __restrict__ thrpart) {
  int bh = blockIdx.x, h = bh & 7;
  int tid = threadIdx.x;
  float local = 0.f;
  for (int e = tid; e < A_ * D_; e += 256) {
    int a = e >> 6, d = e & 63;
    const float* pm = Pm + (size_t)bh * NCHUNK_ * A_ + a;
    float M = -1e30f;
    for (int c = 0; c < NCHUNK_; ++c) M = fmaxf(M, pm[c * A_]);
    float s = 0.f, L = 0.f;
    for (int c = 0; c < NCHUNK_; ++c) {
      float w = __expf(pm[c * A_] - M);
      L += w * Pl[(size_t)bh * NCHUNK_ * A_ + c * A_ + a];
      s += w * Pkv[(((size_t)bh * NCHUNK_ + c) * A_ + a) * D_ + d];
    }
    float val = s / L;
    kv[((size_t)bh * A_ + a) * D_ + d] = val;
    local += val * w_thresh[h * 64 + d];
  }
  __shared__ float red[256];
  red[tid] = local;
  __syncthreads();
  for (int s = 128; s > 0; s >>= 1) {
    if (tid < s) red[tid] += red[tid + s];
    __syncthreads();
  }
  if (tid == 0) thrpart[bh] = red[0];
}

// -------- Kernel 4b: denoise/mask/threshold + softmax_d -> kv2^T bf16 [bh,d,a] --------
__global__ void kv_post(const float* __restrict__ kv, const float* __restrict__ thrpart,
                        const float* __restrict__ w_den, const float* __restrict__ b_den,
                        const float* __restrict__ w_msk, const float* __restrict__ b_msk,
                        const float* __restrict__ b_thresh, __bf16* __restrict__ kv2t) {
  int row = blockIdx.x;  // bh*256 + a
  int bh = row >> 8, a = row & 255;
  int j = threadIdx.x;   // 0..63
  __shared__ float kr[64];
  __shared__ float red[64];
  kr[j] = kv[(size_t)row * 64 + j];
  __syncthreads();
  float ts = 0.f;
  for (int i = 0; i < BH_; ++i) ts += thrpart[i];
  float thr = sigmoidf_(ts / (float)(B_ * A_) + b_thresh[0]);
  float den = 0.f, msk = 0.f;
  for (int i = 0; i < 64; ++i) {
    float kvi = kr[i];
    den += w_den[j * 64 + i] * kvi;
    msk += w_msk[j * 64 + i] * kvi;
  }
  den = sigmoidf_(den + b_den[j]);
  msk = sigmoidf_(msk + b_msk[j]);
  float t = kr[j] * ((msk > thr) ? 1.f : 0.f) + den;
  red[j] = t;
  __syncthreads();
  for (int s = 32; s > 0; s >>= 1) {
    if (j < s) red[j] = fmaxf(red[j], red[j + s]);
    __syncthreads();
  }
  float M = red[0];
  __syncthreads();
  float p = __expf(t - M);
  red[j] = p;
  __syncthreads();
  for (int s = 32; s > 0; s >>= 1) {
    if (j < s) red[j] += red[j + s];
    __syncthreads();
  }
  float S = red[0];
  kv2t[((size_t)bh * 64 + j) * 256 + a] = (__bf16)(p / S);
}

// -------- Kernel 5: fused qa=softmax(q@agent^T * scale); out = qa @ kv2 --------
// grid (N_/32, BH_), block 256 (8 waves); each wave owns a fixed n-tile.
__global__ void attn_out(const __bf16* __restrict__ qb_all, const __bf16* __restrict__ agb,
                         const __bf16* __restrict__ kv2t, float* __restrict__ out) {
  __shared__ __attribute__((aligned(32))) float  sst[32][256];
  __shared__ __attribute__((aligned(32))) __bf16 pst[32][256];
  int bh = blockIdx.y, h = bh & 7, b = bh >> 3;
  int nbase = blockIdx.x * 32;
  int tid = threadIdx.x;
  int wave = tid >> 5, lane = tid & 31, r = lane & 15, hl = lane >> 4;
  const __bf16* qb = qb_all + (size_t)bh * N_ * D_;
  const __bf16* ag = agb + (size_t)h * A_ * D_;
  v8f zero = {0.f, 0.f, 0.f, 0.f, 0.f, 0.f, 0.f, 0.f};
  int ti = wave >> 2, tjbase = (wave & 3) * 4;
  v16bf a0 = frag_ld(qb + (size_t)(nbase + ti * 16) * 64, 64, lane);
  v16bf a1 = frag_ld(qb + (size_t)(nbase + ti * 16) * 64 + 32, 64, lane);
#pragma unroll
  for (int u = 0; u < 4; ++u) {
    int tj = tjbase + u;
    v16bf b0 = frag_ld(ag + (size_t)(tj * 16) * 64, 64, lane);
    v16bf b1 = frag_ld(ag + (size_t)(tj * 16) * 64 + 32, 64, lane);
    v8f s = zero;
    s = wmma_bf16(a0, b0, s);
    s = wmma_bf16(a1, b1, s);
#pragma unroll
    for (int rr = 0; rr < 8; ++rr)
      sst[ti * 16 + rr + 8 * hl][tj * 16 + r] = s[rr] * 0.125f;  // 64^-0.5
  }
  __syncthreads();
  if (tid < 32) {
    float mx = -1e30f;
    for (int c = 0; c < 256; ++c) mx = fmaxf(mx, sst[tid][c]);
    float sum = 0.f;
    for (int c = 0; c < 256; ++c) sum += __expf(sst[tid][c] - mx);
    float inv = 1.f / sum;
    for (int c = 0; c < 256; ++c) pst[tid][c] = (__bf16)(__expf(sst[tid][c] - mx) * inv);
  }
  __syncthreads();
  int oi = wave >> 2, oj = wave & 3;
  v8f acc = zero;
#pragma unroll
  for (int ks = 0; ks < 8; ++ks) {
    v16bf a = frag_ld(&pst[oi * 16][0] + ks * 32, 256, lane);
    v16bf bb = frag_ld(kv2t + ((size_t)bh * 64 + oj * 16) * 256 + ks * 32, 256, lane);
    acc = wmma_bf16(a, bb, acc);
  }
#pragma unroll
  for (int rr = 0; rr < 8; ++rr) {
    int n = nbase + oi * 16 + rr + 8 * hl;
    out[((size_t)b * N_ + n) * DIM_ + h * 64 + oj * 16 + r] = acc[rr];
  }
}

// ---------------- Workspace layout (bytes) ----------------
static constexpr size_t OFF_WQ  = 0;           // 3*512*512 bf16   = 1,572,864
static constexpr size_t OFF_AG  = 1572864;     // 8*256*64 bf16    =   262,144
static constexpr size_t OFF_Q   = 1835008;     // 16*16384*64 bf16 = 33,554,432
static constexpr size_t OFF_K   = 35389440;
static constexpr size_t OFF_VT  = 68943872;
static constexpr size_t OFF_PM  = 102498304;   // 16*32*256 f32    =   524,288
static constexpr size_t OFF_PL  = 103022592;
static constexpr size_t OFF_PKV = 103546880;   // 16*32*256*64 f32 = 33,554,432
static constexpr size_t OFF_KV  = 137101312;   // 16*256*64 f32    =  1,048,576
static constexpr size_t OFF_K2T = 138149888;   // 16*64*256 bf16   =   524,288
static constexpr size_t OFF_TP  = 138674176;   // 16 f32 partials

extern "C" void kernel_launch(void* const* d_in, const int* in_sizes, int n_in,
                              void* d_out, int out_size, void* d_ws, size_t ws_size,
                              hipStream_t stream) {
  const float* x      = (const float*)d_in[0];
  const float* w_qkv  = (const float*)d_in[1];
  const float* agent  = (const float*)d_in[2];
  const float* w_den  = (const float*)d_in[3];
  const float* b_den  = (const float*)d_in[4];
  const float* w_msk  = (const float*)d_in[5];
  const float* b_msk  = (const float*)d_in[6];
  const float* w_thr  = (const float*)d_in[7];
  const float* b_thr  = (const float*)d_in[8];
  float* out = (float*)d_out;

  char* ws = (char*)d_ws;
  __bf16* wqb  = (__bf16*)(ws + OFF_WQ);
  __bf16* agb  = (__bf16*)(ws + OFF_AG);
  __bf16* qb   = (__bf16*)(ws + OFF_Q);
  __bf16* kb   = (__bf16*)(ws + OFF_K);
  __bf16* vtb  = (__bf16*)(ws + OFF_VT);
  float*  Pm   = (float*)(ws + OFF_PM);
  float*  Pl   = (float*)(ws + OFF_PL);
  float*  Pkv  = (float*)(ws + OFF_PKV);
  float*  kv   = (float*)(ws + OFF_KV);
  __bf16* kv2t = (__bf16*)(ws + OFF_K2T);
  float*  tp   = (float*)(ws + OFF_TP);

  cvt_weights<<<3584, 256, 0, stream>>>(w_qkv, agent, wqb, agb);
  qkv_gemm<<<2048, 256, 0, stream>>>(x, wqb, qb, kb, vtb);
  flash_kv<<<dim3(NCHUNK_, BH_), 512, 0, stream>>>(agb, kb, vtb, Pm, Pl, Pkv);
  merge_kv<<<BH_, 256, 0, stream>>>(Pm, Pl, Pkv, w_thr, kv, tp);
  kv_post<<<BH_ * A_, 64, 0, stream>>>(kv, tp, w_den, b_den, w_msk, b_msk, b_thr, kv2t);
  attn_out<<<dim3(N_ / 32, BH_), 256, 0, stream>>>(qb, agb, kv2t, out);
}
